// Transformer_RGBD_11647951306968
// MI455X (gfx1250) — compile-verified, matrix operand order checked by reading, not run on hardware
//
#include <hip/hip_runtime.h>
#include <math.h>

#define D_MODEL 256
#define N_HEADS 8
#define HEAD_DIM 32
#define SEQ 1024
#define BATCH 4
#define FF_DIM 2048
#define NTOK (SEQ * BATCH)            // 4096 token rows (token = s*B + b)
#define LN_EPS 1e-5f
#define QK_SCALE 0.17677669529663687f // 32^-0.5

typedef __attribute__((ext_vector_type(16))) __bf16 v16bf;
typedef __attribute__((ext_vector_type(8)))  __bf16 v8bf;
typedef __attribute__((ext_vector_type(8)))  float  v8f;

__device__ __forceinline__ __bf16 f2bf(float f) {
    unsigned u = __builtin_bit_cast(unsigned, f);
    unsigned r = (u + 0x7FFFu + ((u >> 16) & 1u)) >> 16;   // round-to-nearest-even
    return __builtin_bit_cast(__bf16, (unsigned short)r);
}

// two contiguous 16B runs -> one 16x16x32 WMMA fragment
__device__ __forceinline__ v16bf load_frag(const __bf16* p0, const __bf16* p1) {
    v8bf lo = *(const v8bf*)p0;
    v8bf hi = *(const v8bf*)p1;
    return __builtin_shufflevector(lo, hi, 0, 1, 2, 3, 4, 5, 6, 7,
                                           8, 9, 10, 11, 12, 13, 14, 15);
}

// async global -> LDS 16B copy (per-lane), tracked by ASYNCcnt
__device__ __forceinline__ void async_b128(unsigned lds_addr, const __bf16* gptr) {
    asm volatile("global_load_async_to_lds_b128 %0, %1, off"
                 :: "v"(lds_addr), "v"(gptr) : "memory");
}
__device__ __forceinline__ void wait_async0() {
    asm volatile("s_wait_asynccnt 0x0" ::: "memory");
}
__device__ __forceinline__ unsigned lds_u32(const void* p) {
    return (unsigned)(uintptr_t)p;    // low 32 bits of generic ptr = LDS offset
}

// ---------------------------------------------------------------------------
// out = (relu?)((A[M,K] @ W[N,K]^T + bias[N]) * out_scale)
// A, W bf16 row-major. out_mode: 0 = f32 [M,N], 1 = bf16 [M,N],
// 2 = bf16 value-transposed Vt[(b*D + col)*SEQ + s]  (requires M==NTOK,N==D).
// block = 256 (8 waves). Block tile 64x128 (wave: 16x64). K chunks of 32 are
// double-buffer staged in LDS via async global->LDS; fragments via ds_load_b128.
// grid.x = N/128, grid.y = M/64.
// ---------------------------------------------------------------------------
__global__ __launch_bounds__(256)
void gemm_bf16(const __bf16* __restrict__ A, const __bf16* __restrict__ W,
               const float* __restrict__ bias, void* __restrict__ outp,
               int M, int N, int K, int relu, float out_scale, int out_mode) {
    __shared__ __align__(16) __bf16 Abuf[2][64 * 32];    // 4 KB each
    __shared__ __align__(16) __bf16 Bbuf[2][128 * 32];   // 8 KB each

    const int tid  = threadIdx.x;
    const int lane = tid & 31;
    const int wave = tid >> 5;
    const int l15  = lane & 15;
    const int half = lane >> 4;
    const int wm = wave >> 1;          // 0..3  (M sub-tile)
    const int wn = wave & 1;           // 0..1  (N sub-strip of 64)
    const int tmB = blockIdx.y * 64;
    const int tnB = blockIdx.x * 128;

    // staging assignments: A tile 64x32 (one b128/thread), B tile 128x32 (two)
    const int ar = tid >> 2, aseg = tid & 3;
    const __bf16* aG = A + (size_t)(tmB + ar) * K + aseg * 8;
    const unsigned aL[2] = { lds_u32(&Abuf[0][ar * 32 + aseg * 8]),
                             lds_u32(&Abuf[1][ar * 32 + aseg * 8]) };
    const int br0 = (tid * 2) >> 2,     bs0 = (tid * 2) & 3;
    const int br1 = (tid * 2 + 1) >> 2, bs1 = (tid * 2 + 1) & 3;
    const __bf16* bG0 = W + (size_t)(tnB + br0) * K + bs0 * 8;
    const __bf16* bG1 = W + (size_t)(tnB + br1) * K + bs1 * 8;
    const unsigned bL0[2] = { lds_u32(&Bbuf[0][br0 * 32 + bs0 * 8]),
                              lds_u32(&Bbuf[1][br0 * 32 + bs0 * 8]) };
    const unsigned bL1[2] = { lds_u32(&Bbuf[0][br1 * 32 + bs1 * 8]),
                              lds_u32(&Bbuf[1][br1 * 32 + bs1 * 8]) };

    v8f acc[4] = {v8f{}, v8f{}, v8f{}, v8f{}};

    // prologue: stage chunk 0
    async_b128(aL[0], aG);
    async_b128(bL0[0], bG0);
    async_b128(bL1[0], bG1);
    wait_async0();
    __syncthreads();

    const int nchunks = K >> 5;
    for (int c = 0; c < nchunks; ++c) {
        const int cur = c & 1;
        if (c + 1 < nchunks) {                 // stage next chunk while computing
            const int k1 = (c + 1) * 32;
            async_b128(aL[cur ^ 1], aG + k1);
            async_b128(bL0[cur ^ 1], bG0 + k1);
            async_b128(bL1[cur ^ 1], bG1 + k1);
        }
        const __bf16* aRow = &Abuf[cur][(wm * 16 + l15) * 32];
        const v16bf a = load_frag(aRow + half * 8, aRow + 16 + half * 8);
        #pragma unroll
        for (int j = 0; j < 4; ++j) {
            const __bf16* bRow = &Bbuf[cur][(wn * 64 + j * 16 + l15) * 32];
            const v16bf bf = load_frag(bRow + half * 8, bRow + 16 + half * 8);
            acc[j] = __builtin_amdgcn_wmma_f32_16x16x32_bf16(
                false, a, false, bf, (short)0, acc[j], false, false);
        }
        wait_async0();
        __syncthreads();
    }

    #pragma unroll
    for (int j = 0; j < 4; ++j) {
        const int col = tnB + wn * 64 + j * 16 + l15;
        const float bv = bias ? bias[col] : 0.f;
        #pragma unroll
        for (int r = 0; r < 8; ++r) {
            const int row = tmB + wm * 16 + r + 8 * half;
            float vv = (acc[j][r] + bv) * out_scale;
            if (relu) vv = fmaxf(vv, 0.f);
            if (out_mode == 0) {
                ((float*)outp)[(size_t)row * N + col] = vv;
            } else if (out_mode == 1) {
                ((__bf16*)outp)[(size_t)row * N + col] = f2bf(vv);
            } else {
                const int s  = row >> 2;       // token -> (s, b)
                const int bb = row & 3;
                ((__bf16*)outp)[((size_t)bb * D_MODEL + col) * SEQ + s] = f2bf(vv);
            }
        }
    }
}

// ---------------------------------------------------------------------------
// Fused dual-branch flash attention. One block = 4 waves = 4 query tiles of the
// SAME (b,h): the 32-key K/Kd/V/Vd chunks are async-staged into LDS once per
// block and consumed by all 4 waves.
//   P1 = softmax(q k^T), P2 = softmax(qd kd^T)  (scale folded into q/qd)
//   o_r = (1-a) P1@V + a P2@V ;  o_d = (1-b) P2@Vd + b P1@Vd
// q/k/qd/kd bf16 token-major [NTOK, D]; vt/vdt bf16 Vt[(b*D+col)*SEQ+s].
// grid.x = B*H*(SEQ/64) = 512, block = 128.
// ---------------------------------------------------------------------------
__global__ __launch_bounds__(128)
void flash_fused(const __bf16* __restrict__ q,  const __bf16* __restrict__ k,
                 const __bf16* __restrict__ qd, const __bf16* __restrict__ kd,
                 const __bf16* __restrict__ vt, const __bf16* __restrict__ vdt,
                 __bf16* __restrict__ o_r, __bf16* __restrict__ o_d,
                 const float* __restrict__ alpha_p, const float* __restrict__ beta_p) {
    __shared__ __align__(16) __bf16 Kbuf[2][32 * 32];   // keys x hd
    __shared__ __align__(16) __bf16 Kdbuf[2][32 * 32];
    __shared__ __align__(16) __bf16 Vbuf[2][32 * 32];   // hd x keys (from Vt)
    __shared__ __align__(16) __bf16 Vdbuf[2][32 * 32];
    __shared__ __align__(16) __bf16 pbuf[4][2][16 * 32];

    const int tid  = threadIdx.x;
    const int lane = tid & 31;
    const int wave = tid >> 5;
    const int l15  = lane & 15;
    const int half = lane >> 4;

    const int bh = blockIdx.x >> 4;            // 0..31
    const int qt = (blockIdx.x & 15) * 4 + wave;
    const int b  = bh >> 3;
    const int h  = bh & 7;

    const float alpha = *alpha_p, beta = *beta_p;

    // Q fragments (A layout), loaded once (K-dim == HD == 32)
    const __bf16* qrow  = q  + ((size_t)(qt * 16 + l15) * BATCH + b) * D_MODEL + h * HEAD_DIM;
    const __bf16* qdrow = qd + ((size_t)(qt * 16 + l15) * BATCH + b) * D_MODEL + h * HEAD_DIM;
    const v16bf aq1 = load_frag(qrow  + half * 8, qrow  + 16 + half * 8);
    const v16bf aq2 = load_frag(qdrow + half * 8, qdrow + 16 + half * 8);

    // staging assignment: each tile 32 rows x 64B; thread -> (row, 16B seg)
    const int sr = tid >> 2, sseg = tid & 3;
    const size_t khead = (size_t)sr * BATCH * D_MODEL + (size_t)b * D_MODEL + h * HEAD_DIM + sseg * 8;
    const __bf16* kG  = k  + khead;    // + kt*BATCH*D_MODEL per chunk
    const __bf16* kdG = kd + khead;
    const size_t vhead = ((size_t)b * D_MODEL + h * HEAD_DIM + sr) * SEQ + sseg * 8;
    const __bf16* vG  = vt  + vhead;   // + kt per chunk
    const __bf16* vdG = vdt + vhead;
    const unsigned ldsoff = (unsigned)(sr * 32 + sseg * 8);
    const unsigned kL[2]  = { lds_u32(&Kbuf[0][0])  + ldsoff * 2, lds_u32(&Kbuf[1][0])  + ldsoff * 2 };
    const unsigned kdL[2] = { lds_u32(&Kdbuf[0][0]) + ldsoff * 2, lds_u32(&Kdbuf[1][0]) + ldsoff * 2 };
    const unsigned vL[2]  = { lds_u32(&Vbuf[0][0])  + ldsoff * 2, lds_u32(&Vbuf[1][0])  + ldsoff * 2 };
    const unsigned vdL[2] = { lds_u32(&Vdbuf[0][0]) + ldsoff * 2, lds_u32(&Vdbuf[1][0]) + ldsoff * 2 };

    float M1[8], L1[8], M2[8], L2[8];
    #pragma unroll
    for (int r = 0; r < 8; ++r) { M1[r] = -1e30f; L1[r] = 0.f; M2[r] = -1e30f; L2[r] = 0.f; }
    v8f a1v[2] = {v8f{}, v8f{}}, a1w[2] = {v8f{}, v8f{}};
    v8f a2v[2] = {v8f{}, v8f{}}, a2w[2] = {v8f{}, v8f{}};

    // prologue: stage chunk 0
    async_b128(kL[0],  kG);
    async_b128(kdL[0], kdG);
    async_b128(vL[0],  vG);
    async_b128(vdL[0], vdG);
    wait_async0();
    __syncthreads();

    for (int c = 0; c < SEQ / 32; ++c) {
        const int cur = c & 1;
        if (c + 1 < SEQ / 32) {
            const int kt1 = (c + 1) * 32;
            async_b128(kL[cur ^ 1],  kG  + (size_t)kt1 * BATCH * D_MODEL);
            async_b128(kdL[cur ^ 1], kdG + (size_t)kt1 * BATCH * D_MODEL);
            async_b128(vL[cur ^ 1],  vG  + kt1);
            async_b128(vdL[cur ^ 1], vdG + kt1);
        }

        // ---- scores from LDS K tiles: 2 tiles per branch
        v8f s0 = {}, s1 = {}, t0 = {}, t1 = {};
        {
            const __bf16* kr0 = &Kbuf[cur][(size_t)l15 * 32];
            const __bf16* kr1 = &Kbuf[cur][(size_t)(16 + l15) * 32];
            v16bf bk = load_frag(kr0 + half * 8, kr0 + 16 + half * 8);
            s0 = __builtin_amdgcn_wmma_f32_16x16x32_bf16(false, aq1, false, bk, (short)0, s0, false, false);
            bk = load_frag(kr1 + half * 8, kr1 + 16 + half * 8);
            s1 = __builtin_amdgcn_wmma_f32_16x16x32_bf16(false, aq1, false, bk, (short)0, s1, false, false);
            const __bf16* dr0 = &Kdbuf[cur][(size_t)l15 * 32];
            const __bf16* dr1 = &Kdbuf[cur][(size_t)(16 + l15) * 32];
            bk = load_frag(dr0 + half * 8, dr0 + 16 + half * 8);
            t0 = __builtin_amdgcn_wmma_f32_16x16x32_bf16(false, aq2, false, bk, (short)0, t0, false, false);
            bk = load_frag(dr1 + half * 8, dr1 + 16 + half * 8);
            t1 = __builtin_amdgcn_wmma_f32_16x16x32_bf16(false, aq2, false, bk, (short)0, t1, false, false);
        }

        // ---- online softmax, both branches
        float c1[8], c2[8];
        #pragma unroll
        for (int r = 0; r < 8; ++r) {
            float m = fmaxf(s0[r], s1[r]);
            #pragma unroll
            for (int off = 1; off < 16; off <<= 1) m = fmaxf(m, __shfl_xor(m, off, 32));
            const float nm = fmaxf(M1[r], m);
            c1[r] = __expf(M1[r] - nm); M1[r] = nm;
            const float p0 = __expf(s0[r] - nm), p1 = __expf(s1[r] - nm);
            s0[r] = p0; s1[r] = p1;
            float rs = p0 + p1;
            #pragma unroll
            for (int off = 1; off < 16; off <<= 1) rs += __shfl_xor(rs, off, 32);
            L1[r] = L1[r] * c1[r] + rs;
        }
        #pragma unroll
        for (int r = 0; r < 8; ++r) {
            float m = fmaxf(t0[r], t1[r]);
            #pragma unroll
            for (int off = 1; off < 16; off <<= 1) m = fmaxf(m, __shfl_xor(m, off, 32));
            const float nm = fmaxf(M2[r], m);
            c2[r] = __expf(M2[r] - nm); M2[r] = nm;
            const float p0 = __expf(t0[r] - nm), p1 = __expf(t1[r] - nm);
            t0[r] = p0; t1[r] = p1;
            float rs = p0 + p1;
            #pragma unroll
            for (int off = 1; off < 16; off <<= 1) rs += __shfl_xor(rs, off, 32);
            L2[r] = L2[r] * c2[r] + rs;
        }
        #pragma unroll
        for (int r = 0; r < 8; ++r) {
            a1v[0][r] *= c1[r]; a1v[1][r] *= c1[r]; a1w[0][r] *= c1[r]; a1w[1][r] *= c1[r];
            a2v[0][r] *= c2[r]; a2v[1][r] *= c2[r]; a2w[0][r] *= c2[r]; a2w[1][r] *= c2[r];
        }

        // ---- P tiles: C-layout -> A-layout through per-wave LDS
        asm volatile("s_wait_dscnt 0x0" ::: "memory");
        {
            __bf16* pw1 = &pbuf[wave][0][0];
            __bf16* pw2 = &pbuf[wave][1][0];
            #pragma unroll
            for (int r = 0; r < 8; ++r) {
                const int row = r + 8 * half;
                pw1[row * 32 + l15]      = f2bf(s0[r]);
                pw1[row * 32 + 16 + l15] = f2bf(s1[r]);
                pw2[row * 32 + l15]      = f2bf(t0[r]);
                pw2[row * 32 + 16 + l15] = f2bf(t1[r]);
            }
        }
        asm volatile("s_wait_dscnt 0x0" ::: "memory");
        const __bf16* pr1 = &pbuf[wave][0][l15 * 32];
        const __bf16* pr2 = &pbuf[wave][1][l15 * 32];
        const v16bf pa1 = load_frag(pr1 + half * 8, pr1 + 16 + half * 8);
        const v16bf pa2 = load_frag(pr2 + half * 8, pr2 + 16 + half * 8);

        // ---- P @ V from shared LDS V tiles (8 WMMAs)
        #pragma unroll
        for (int t = 0; t < 2; ++t) {
            const __bf16* vrow = &Vbuf[cur][(size_t)(t * 16 + l15) * 32];
            const v16bf bv = load_frag(vrow + half * 8, vrow + 16 + half * 8);
            a1v[t] = __builtin_amdgcn_wmma_f32_16x16x32_bf16(false, pa1, false, bv, (short)0, a1v[t], false, false);
            a2v[t] = __builtin_amdgcn_wmma_f32_16x16x32_bf16(false, pa2, false, bv, (short)0, a2v[t], false, false);
            const __bf16* wrow = &Vdbuf[cur][(size_t)(t * 16 + l15) * 32];
            const v16bf bw = load_frag(wrow + half * 8, wrow + 16 + half * 8);
            a1w[t] = __builtin_amdgcn_wmma_f32_16x16x32_bf16(false, pa1, false, bw, (short)0, a1w[t], false, false);
            a2w[t] = __builtin_amdgcn_wmma_f32_16x16x32_bf16(false, pa2, false, bw, (short)0, a2w[t], false, false);
        }

        wait_async0();
        __syncthreads();
    }

    // ---- epilogue: normalize, alpha/beta mix, store bf16 token-major
    const float wa1 = 1.f - alpha, wa2 = alpha;
    const float wb2 = 1.f - beta,  wb1 = beta;
    #pragma unroll
    for (int r = 0; r < 8; ++r) {
        const int s = qt * 16 + r + 8 * half;
        const float iL1 = 1.f / L1[r], iL2 = 1.f / L2[r];
        const size_t base = ((size_t)s * BATCH + b) * D_MODEL + h * HEAD_DIM;
        o_r[base + l15]      = f2bf(wa1 * a1v[0][r] * iL1 + wa2 * a2v[0][r] * iL2);
        o_r[base + 16 + l15] = f2bf(wa1 * a1v[1][r] * iL1 + wa2 * a2v[1][r] * iL2);
        o_d[base + l15]      = f2bf(wb2 * a2w[0][r] * iL2 + wb1 * a1w[0][r] * iL1);
        o_d[base + 16 + l15] = f2bf(wb2 * a2w[1][r] * iL2 + wb1 * a1w[1][r] * iL1);
    }
}

// ---------------------------------------------------------------------------
__global__ __launch_bounds__(256)
void layernorm_res(const float* __restrict__ x, const float* __restrict__ res,
                   const float* __restrict__ g, const float* __restrict__ bb,
                   float* __restrict__ outf, __bf16* __restrict__ outb, int rows) {
    const int row  = blockIdx.x * 8 + (threadIdx.x >> 5);
    const int lane = threadIdx.x & 31;
    if (row >= rows) return;
    const float* xr = x + (size_t)row * D_MODEL;
    const float* rr = res + (size_t)row * D_MODEL;
    float vals[8], s = 0.f, s2 = 0.f;
    #pragma unroll
    for (int i = 0; i < 8; ++i) {
        const float v = xr[i * 32 + lane] + rr[i * 32 + lane];
        vals[i] = v; s += v; s2 += v * v;
    }
    #pragma unroll
    for (int off = 1; off < 32; off <<= 1) {
        s  += __shfl_xor(s,  off, 32);
        s2 += __shfl_xor(s2, off, 32);
    }
    const float mean = s * (1.f / D_MODEL);
    const float var  = s2 * (1.f / D_MODEL) - mean * mean;
    const float rstd = rsqrtf(var + LN_EPS);
    #pragma unroll
    for (int i = 0; i < 8; ++i) {
        const int c = i * 32 + lane;
        const float o = (vals[i] - mean) * rstd * g[c] + bb[c];
        outf[(size_t)row * D_MODEL + c] = o;
        outb[(size_t)row * D_MODEL + c] = f2bf(o);
    }
}

__global__ void add_bf(const float* __restrict__ a, const float* __restrict__ b,
                       __bf16* __restrict__ o, int n) {
    const int i = (blockIdx.x * blockDim.x + threadIdx.x) * 4;
    if (i < n) {
        const float4 x = *(const float4*)(a + i);
        const float4 y = *(const float4*)(b + i);
        o[i]     = f2bf(x.x + y.x);
        o[i + 1] = f2bf(x.y + y.y);
        o[i + 2] = f2bf(x.z + y.z);
        o[i + 3] = f2bf(x.w + y.w);
    }
}

__global__ void conv_bf(const float* __restrict__ in, __bf16* __restrict__ out, int n) {
    const int i = (blockIdx.x * blockDim.x + threadIdx.x) * 4;
    if (i < n) {
        const float4 x = *(const float4*)(in + i);
        out[i]     = f2bf(x.x);
        out[i + 1] = f2bf(x.y);
        out[i + 2] = f2bf(x.z);
        out[i + 3] = f2bf(x.w);
    }
}

// ---------------------------------------------------------------------------
extern "C" void kernel_launch(void* const* d_in, const int* in_sizes, int n_in,
                              void* d_out, int out_size, void* d_ws, size_t ws_size,
                              hipStream_t stream) {
    (void)in_sizes; (void)n_in; (void)out_size; (void)ws_size;
    const float* src  = (const float*)d_in[0];
    const float* srcd = (const float*)d_in[1];
    const float* pos  = (const float*)d_in[2];
    const float* posd = (const float*)d_in[3];

    float* ws = (float*)d_ws;
    const size_t T = (size_t)NTOK * D_MODEL;   // 1,048,576 elements

    float* cur_r = ws + 0 * T;
    float* cur_d = ws + 1 * T;
    float* t1    = ws + 2 * T;
    float* t2    = ws + 3 * T;
    float* xr    = ws + 4 * T;
    float* xd    = ws + 5 * T;
    __bf16* bfb      = (__bf16*)(ws + 6 * T);
    __bf16* qp_bf    = bfb + 0 * T;
    __bf16* qpd_bf   = bfb + 1 * T;
    __bf16* q_bf     = bfb + 2 * T;
    __bf16* k_bf     = bfb + 3 * T;
    __bf16* vt_bf    = bfb + 4 * T;
    __bf16* qd_bf    = bfb + 5 * T;
    __bf16* kd_bf    = bfb + 6 * T;
    __bf16* vdt_bf   = bfb + 7 * T;
    __bf16* or_bf    = bfb + 8 * T;
    __bf16* od_bf    = bfb + 9 * T;
    __bf16* xr_bf    = bfb + 10 * T;
    __bf16* xd_bf    = bfb + 11 * T;
    __bf16* curr_bf  = bfb + 12 * T;
    __bf16* curd_bf  = bfb + 13 * T;
    __bf16* ffn_bf   = bfb + 14 * T;          // 8*T elements
    __bf16* wbuf     = bfb + 22 * T;
    const size_t wIN  = 768 * 256;
    const size_t wSQ  = 256 * 256;
    const size_t wFF  = 2048 * 256;
    __bf16* w_inR  = wbuf;
    __bf16* w_outR = w_inR + wIN;
    __bf16* w_inD  = w_outR + wSQ;
    __bf16* w_outD = w_inD + wIN;
    __bf16* w_l1   = w_outD + wSQ;
    __bf16* w_l2   = w_l1 + wFF;
    __bf16* w_l1d  = w_l2 + wFF;
    __bf16* w_l2d  = w_l1d + wFF;

    hipMemcpyAsync(cur_r, src,  T * sizeof(float), hipMemcpyDeviceToDevice, stream);
    hipMemcpyAsync(cur_d, srcd, T * sizeof(float), hipMemcpyDeviceToDevice, stream);
    conv_bf<<<(int)(T / 1024), 256, 0, stream>>>(src,  curr_bf, (int)T);
    conv_bf<<<(int)(T / 1024), 256, 0, stream>>>(srcd, curd_bf, (int)T);

    const dim3 blk(256);
    const dim3 gProj(D_MODEL / 128, NTOK / 64);      // (2, 64)
    const dim3 gFF1(FF_DIM / 128, NTOK / 64);        // (16, 64)

    for (int l = 0; l < 6; ++l) {
        const float* const* P = (const float* const*)(d_in + 4 + l * 26);
        const float *rgb_in_w = P[0],  *rgb_in_b = P[1];
        const float *rgb_out_w= P[2],  *rgb_out_b= P[3];
        const float *dpt_in_w = P[4],  *dpt_in_b = P[5];
        const float *dpt_out_w= P[6],  *dpt_out_b= P[7];
        const float *alpha    = P[8],  *beta     = P[9];
        const float *lin1_w = P[10], *lin1_b = P[11], *lin2_w = P[12], *lin2_b = P[13];
        const float *n1g = P[14], *n1b = P[15], *n2g = P[16], *n2b = P[17];
        const float *lin1d_w = P[18], *lin1d_b = P[19], *lin2d_w = P[20], *lin2d_b = P[21];
        const float *n1dg = P[22], *n1db = P[23], *n2dg = P[24], *n2db = P[25];

        conv_bf<<<(int)(wIN / 1024), 256, 0, stream>>>(rgb_in_w,  w_inR,  (int)wIN);
        conv_bf<<<(int)(wSQ / 1024), 256, 0, stream>>>(rgb_out_w, w_outR, (int)wSQ);
        conv_bf<<<(int)(wIN / 1024), 256, 0, stream>>>(dpt_in_w,  w_inD,  (int)wIN);
        conv_bf<<<(int)(wSQ / 1024), 256, 0, stream>>>(dpt_out_w, w_outD, (int)wSQ);
        conv_bf<<<(int)(wFF / 1024), 256, 0, stream>>>(lin1_w,  w_l1,  (int)wFF);
        conv_bf<<<(int)(wFF / 1024), 256, 0, stream>>>(lin2_w,  w_l2,  (int)wFF);
        conv_bf<<<(int)(wFF / 1024), 256, 0, stream>>>(lin1d_w, w_l1d, (int)wFF);
        conv_bf<<<(int)(wFF / 1024), 256, 0, stream>>>(lin2d_w, w_l2d, (int)wFF);

        add_bf<<<(int)(T / 1024), 256, 0, stream>>>(cur_r, pos,  qp_bf,  (int)T);
        add_bf<<<(int)(T / 1024), 256, 0, stream>>>(cur_d, posd, qpd_bf, (int)T);

        gemm_bf16<<<gProj, blk, 0, stream>>>(qp_bf,   w_inR,            rgb_in_b,       q_bf,   NTOK, D_MODEL, D_MODEL, 0, QK_SCALE, 1);
        gemm_bf16<<<gProj, blk, 0, stream>>>(qp_bf,   w_inR + wSQ,      rgb_in_b + 256, k_bf,   NTOK, D_MODEL, D_MODEL, 0, 1.f, 1);
        gemm_bf16<<<gProj, blk, 0, stream>>>(curr_bf, w_inR + 2 * wSQ,  rgb_in_b + 512, vt_bf,  NTOK, D_MODEL, D_MODEL, 0, 1.f, 2);
        gemm_bf16<<<gProj, blk, 0, stream>>>(qpd_bf,  w_inD,            dpt_in_b,       qd_bf,  NTOK, D_MODEL, D_MODEL, 0, QK_SCALE, 1);
        gemm_bf16<<<gProj, blk, 0, stream>>>(qpd_bf,  w_inD + wSQ,      dpt_in_b + 256, kd_bf,  NTOK, D_MODEL, D_MODEL, 0, 1.f, 1);
        gemm_bf16<<<gProj, blk, 0, stream>>>(curd_bf, w_inD + 2 * wSQ,  dpt_in_b + 512, vdt_bf, NTOK, D_MODEL, D_MODEL, 0, 1.f, 2);

        flash_fused<<<512, dim3(128), 0, stream>>>(q_bf, k_bf, qd_bf, kd_bf, vt_bf, vdt_bf,
                                                   or_bf, od_bf, alpha, beta);

        gemm_bf16<<<gProj, blk, 0, stream>>>(or_bf, w_outR, rgb_out_b, t1, NTOK, D_MODEL, D_MODEL, 0, 1.f, 0);
        gemm_bf16<<<gProj, blk, 0, stream>>>(od_bf, w_outD, dpt_out_b, t2, NTOK, D_MODEL, D_MODEL, 0, 1.f, 0);

        layernorm_res<<<NTOK / 8, 256, 0, stream>>>(t1, cur_r, n1g,  n1b,  xr, xr_bf, NTOK);
        layernorm_res<<<NTOK / 8, 256, 0, stream>>>(t2, cur_d, n1dg, n1db, xd, xd_bf, NTOK);

        gemm_bf16<<<gFF1,  blk, 0, stream>>>(xr_bf,  w_l1, lin1_b, ffn_bf, NTOK, FF_DIM,  D_MODEL, 1, 1.f, 1);
        gemm_bf16<<<gProj, blk, 0, stream>>>(ffn_bf, w_l2, lin2_b, t1,     NTOK, D_MODEL, FF_DIM,  0, 1.f, 0);
        layernorm_res<<<NTOK / 8, 256, 0, stream>>>(t1, xr, n2g, n2b, cur_r, curr_bf, NTOK);

        gemm_bf16<<<gFF1,  blk, 0, stream>>>(xd_bf,  w_l1d, lin1d_b, ffn_bf, NTOK, FF_DIM,  D_MODEL, 1, 1.f, 1);
        gemm_bf16<<<gProj, blk, 0, stream>>>(ffn_bf, w_l2d, lin2d_b, t2,     NTOK, D_MODEL, FF_DIM,  0, 1.f, 0);
        layernorm_res<<<NTOK / 8, 256, 0, stream>>>(t2, xd, n2dg, n2db, cur_d, curd_bf, NTOK);
    }

    hipMemcpyAsync((float*)d_out,     cur_r, T * sizeof(float), hipMemcpyDeviceToDevice, stream);
    hipMemcpyAsync((float*)d_out + T, cur_d, T * sizeof(float), hipMemcpyDeviceToDevice, stream);
}
